// KANLikeRegressor_15212774163194
// MI455X (gfx1250) — matrix operand order repaired and checked by reading, not compiled, not run on hardware
//
#include <hip/hip_runtime.h>

// KAN-like regressor, fused WMMA kernel for gfx1250 (MI455X).
// out[n] = sum_d ( relu( relu(x[n,d]*W1[d]+b1[d]) @ W2[d] + b2[d] ) . W3[d] + b3[d] )
// N=65536 rows, D=32 features, H=128 hidden. GEMM via v_wmma_f32_16x16x32_f16.
// Per-wave tile: 32 rows x 64 cols (2x4 fragments), K software-pipelined.

#define TM 128   // rows per block
#define HD 128   // hidden width
#define DF 32    // features
#define AS 136   // LDS stride (halves) for A tile: 272B rows, 16B-aligned, bank-spread
#define BS 136   // LDS stride (halves) for B^T tile (ws uses identical layout)

typedef __attribute__((ext_vector_type(16))) _Float16 v16h;
typedef __attribute__((ext_vector_type(8)))  _Float16 v8h;
typedef __attribute__((ext_vector_type(8)))  float    v8f;

union Frag  { v16h v; v8h h[2]; };
union Chunk { v8h h; float4 f; };

// ---------------- prep: W2 [D,H,H] f32 -> W2T f16, layout [d][col*BS + k] ----------------
__global__ __launch_bounds__(256) void kan_prep_w2t(
    const float* __restrict__ W2, _Float16* __restrict__ W2T)
{
  const int d  = blockIdx.x;
  const int kk = threadIdx.x & (HD - 1);
  const int ch = threadIdx.x >> 7;              // 0..1
  const float* W2d = W2 + (size_t)d * HD * HD;
  _Float16*    dst = W2T + (size_t)d * HD * BS;
  for (int i = 0; i < HD / 2; ++i) {
    const int c = ch + 2 * i;
    dst[c * BS + kk] = (_Float16)W2d[kk * HD + c];   // coalesced b16 stores along kk
  }
}

// ---------------- main fused kernel ----------------
__global__ __launch_bounds__(256) void kan_fused_kernel(
    const float* __restrict__ x,
    const float* __restrict__ W1, const float* __restrict__ b1,
    const float* __restrict__ W2, const float* __restrict__ b2,
    const float* __restrict__ W3, const float* __restrict__ b3,
    const _Float16* __restrict__ W2T, float* __restrict__ out, int usePre)
{
  __shared__ __align__(16) _Float16 Ash[TM * AS];   // h1 tile, row-major [row][k]
  __shared__ __align__(16) _Float16 Bsh[HD * BS];   // W2[d]^T, [col][k]
  __shared__ float xs[TM * DF];                     // x tile (all features)

  const int tid  = threadIdx.x;
  const int lane = tid & 31;
  const int wave = tid >> 5;
  const int rowbase = blockIdx.x * TM;

  // Wave grid over the 128x128 output tile: 4 waves along M, 2 along N.
  const int mw = wave & 3;              // rows mw*32 .. mw*32+31
  const int nw = wave >> 2;             // cols nw*64 .. nw*64+63

  // A-build decomposition: 8 consecutive columns per thread, 8 rows (stride 16)
  const int c8 = (tid & 15) * 8;
  const int rg = tid >> 4;              // 0..15

  // Stage the whole x tile once (contiguous, coalesced: x is [N,D] row-major).
  for (int i = tid; i < TM * DF; i += 256)
    xs[i] = x[(size_t)rowbase * DF + i];

  float rowacc[2][8];
  #pragma unroll
  for (int m = 0; m < 2; ++m)
    #pragma unroll
    for (int j = 0; j < 8; ++j) rowacc[m][j] = 0.f;
  float b3sum = 0.f;

  const int half = lane >> 4;           // 0 = lanes 0..15, 1 = lanes 16..31
  const int l15  = lane & 15;
  const int m0row = mw * 32 + l15;      // A row for M-frag 0 (M-frag 1: +16)

  typedef float f8v __attribute__((ext_vector_type(8)));
  const v8h zero8 = {};
  const int NCHUNK = (HD * BS) / 8;     // 2176 b128 chunks in the B tile

  auto loadA = [&](Frag& fr, int row, int k) {
    const _Float16* p = &Ash[row * AS + k];
    fr.h[0] = *(const v8h*)(p);
    fr.h[1] = *(const v8h*)(p + 16);
  };
  auto loadB = [&](Frag& fr, int col, int k) {
    const _Float16* p = &Bsh[col * BS + k];
    fr.h[0] = *(const v8h*)(p);
    fr.h[1] = *(const v8h*)(p + 16);
  };

  for (int d = 0; d < DF; ++d) {
    __syncthreads();  // previous iteration's LDS reads done (also fences xs on d=0)

    if (usePre) {
      // Issue B-tile global loads first so L2 latency overlaps the A build.
      Chunk breg[9];
      const float4* bsrc = (const float4*)(W2T + (size_t)d * HD * BS);
      #pragma unroll
      for (int i = 0; i < 9; ++i) {
        const int chunk = tid + 256 * i;
        if (chunk < NCHUNK) breg[i].f = bsrc[chunk];
      }

      // ---- Build A = relu(x[:,d]*W1[d,:] + b1[d,:]) with packed f16 math ----
      const v8h w1h = __builtin_convertvector(*(const f8v*)(W1 + d * HD + c8), v8h);
      const v8h b1h = __builtin_convertvector(*(const f8v*)(b1 + d * HD + c8), v8h);
      #pragma unroll
      for (int i = 0; i < 8; ++i) {
        const int r = rg + 16 * i;
        const _Float16 xh = (_Float16)xs[r * DF + d];
        v8h h1 = w1h * xh + b1h;                       // v_pk_fma_f16
        h1 = __builtin_elementwise_max(h1, zero8);     // v_pk_max (relu)
        *(v8h*)(&Ash[r * AS + c8]) = h1;               // one b128 store per row
      }

      // ---- Drain B tile into LDS (straight copy, pads included/never read) ----
      float4* bdst = (float4*)Bsh;
      #pragma unroll
      for (int i = 0; i < 9; ++i) {
        const int chunk = tid + 256 * i;
        if (chunk < NCHUNK) bdst[chunk] = breg[i].f;
      }
    } else {
      // Fallback (workspace too small): convert in-kernel.
      const v8h w1h = __builtin_convertvector(*(const f8v*)(W1 + d * HD + c8), v8h);
      const v8h b1h = __builtin_convertvector(*(const f8v*)(b1 + d * HD + c8), v8h);
      #pragma unroll
      for (int i = 0; i < 8; ++i) {
        const int r = rg + 16 * i;
        const _Float16 xh = (_Float16)xs[r * DF + d];
        v8h h1 = w1h * xh + b1h;
        h1 = __builtin_elementwise_max(h1, zero8);
        *(v8h*)(&Ash[r * AS + c8]) = h1;
      }
      const int cc = tid & (HD - 1);
      const int r2 = tid >> 7;
      const float* W2d = W2 + (size_t)d * HD * HD;
      #pragma unroll 8
      for (int i = 0; i < HD / 2; ++i) {
        const int kk = 2 * i + r2;
        Bsh[cc * BS + kk] = (_Float16)W2d[kk * HD + cc];
      }
    }
    __syncthreads();

    // ---- GEMM: 2x4 fragments per wave, K-steps software-pipelined ----
    v8f acc[2][4] = {};
    Frag ca[2], cb[4], na[2], nb[4];

    loadA(ca[0], m0row,      half * 8);
    loadA(ca[1], m0row + 16, half * 8);
    #pragma unroll
    for (int f = 0; f < 4; ++f) loadB(cb[f], nw * 64 + f * 16 + l15, half * 8);

    #pragma unroll
    for (int ks = 0; ks < 4; ++ks) {
      if (ks < 3) {
        const int kA = (ks + 1) * 32 + half * 8;
        const int kB = (ks + 1) * 32 + half * 8;
        loadA(na[0], m0row,      kA);
        loadA(na[1], m0row + 16, kA);
        #pragma unroll
        for (int f = 0; f < 4; ++f) loadB(nb[f], nw * 64 + f * 16 + l15, kB);
      }
      #pragma unroll
      for (int m = 0; m < 2; ++m)
        #pragma unroll
        for (int f = 0; f < 4; ++f)
          acc[m][f] = __builtin_amdgcn_wmma_f32_16x16x32_f16(
              false, ca[m].v, false, cb[f].v, (short)0, acc[m][f], false, false);
      if (ks < 3) {
        ca[0] = na[0]; ca[1] = na[1];
        #pragma unroll
        for (int f = 0; f < 4; ++f) cb[f] = nb[f];
      }
    }

    // ---- Epilogue: relu(acc + b2) . W3, accumulate lane-partials (reduce later) ----
    // C layout: acc[m][f] VGPR j -> row mw*32 + m*16 + half*8 + j, col nw*64 + f*16 + l15.
    #pragma unroll
    for (int f = 0; f < 4; ++f) {
      const int col = nw * 64 + f * 16 + l15;
      const float bb2 = b2[d * HD + col];
      const float w3v = W3[d * HD + col];
      #pragma unroll
      for (int m = 0; m < 2; ++m)
        #pragma unroll
        for (int j = 0; j < 8; ++j) {
          const float h2 = fmaxf(acc[m][f][j] + bb2, 0.f);
          rowacc[m][j] = fmaf(h2, w3v, rowacc[m][j]);
        }
    }
    b3sum += b3[d];
  }

  // ---- Final reduction: butterfly within 16-lane halves, then across the 2 col-waves ----
  __syncthreads();                      // done reading Ash/Bsh; reuse Ash as scratch
  float* red = (float*)Ash;             // red[nw*128 + row_in_block]

  #pragma unroll
  for (int m = 0; m < 2; ++m) {
    #pragma unroll
    for (int j = 0; j < 8; ++j) {
      float v = rowacc[m][j];
      v += __shfl_xor(v, 1, 32);        // masks < 16 keep the reduction within each half
      v += __shfl_xor(v, 2, 32);
      v += __shfl_xor(v, 4, 32);
      v += __shfl_xor(v, 8, 32);
      rowacc[m][j] = v;
    }
    if (l15 < 8) {
      float v = rowacc[m][0];
      #pragma unroll
      for (int j = 1; j < 8; ++j)
        if (l15 == j) v = rowacc[m][j];
      const int r = mw * 32 + m * 16 + half * 8 + l15;
      red[nw * TM + r] = v;
    }
  }
  __syncthreads();

  if (tid < TM)
    out[(size_t)rowbase + tid] = red[tid] + red[TM + tid] + b3sum;
}

extern "C" void kernel_launch(void* const* d_in, const int* in_sizes, int n_in,
                              void* d_out, int out_size, void* d_ws, size_t ws_size,
                              hipStream_t stream) {
  (void)n_in; (void)out_size;
  const float* x  = (const float*)d_in[0];
  const float* W1 = (const float*)d_in[1];
  const float* b1 = (const float*)d_in[2];
  const float* W2 = (const float*)d_in[3];
  const float* b2 = (const float*)d_in[4];
  const float* W3 = (const float*)d_in[5];
  const float* b3 = (const float*)d_in[6];
  float* out = (float*)d_out;

  const size_t wsNeed = (size_t)DF * HD * BS * sizeof(_Float16);  // ~1.06 MB
  const int usePre = (d_ws != nullptr && ws_size >= wsNeed) ? 1 : 0;
  _Float16* W2T = (_Float16*)d_ws;

  if (usePre) {
    kan_prep_w2t<<<dim3(DF), dim3(256), 0, stream>>>(W2, W2T);
  }

  const int N = in_sizes[0] / DF;        // 65536
  dim3 grid(N / TM);                      // 512 blocks
  dim3 block(256);                        // 8 wave32s
  kan_fused_kernel<<<grid, block, 0, stream>>>(x, W1, b1, W2, b2, W3, b3, W2T, out, usePre);
}
